// GibbsSCVI_20461224198820
// MI455X (gfx1250) — matrix operand order repaired and checked by reading, not compiled, not run on hardware
//
#include <hip/hip_runtime.h>
#include <hip/hip_bf16.h>

// ---------------------------------------------------------------------------
// scVI-style VAE forward (-ELBO scalar) for MI455X / gfx1250 (wave32, WMMA).
// ---------------------------------------------------------------------------

typedef __attribute__((ext_vector_type(16))) _Float16 v16h;
typedef __attribute__((ext_vector_type(8)))  float    v8f;
typedef __attribute__((ext_vector_type(4)))  float    vf4;

union Frag {
  v16h v;
  vf4  f[2];
};

#define BB      1024
#define N_IN    1000
#define N_INP   1008   /* 1000 padded to 63*16 */
#define HH      128
#define LL      32
#define LN_EPS  1e-6f
#define BN_EPS  1e-3f

// ------------------------------ wave reduce --------------------------------
__device__ __forceinline__ float wave_sum(float v) {
#pragma unroll
  for (int o = 16; o > 0; o >>= 1) v += __shfl_xor(v, o, 32);
  return v;
}

// ------------------------------ mask kernels -------------------------------
__global__ void k_mask_init(float* mask) {
  int i = blockIdx.x * blockDim.x + threadIdx.x;
  if (i < N_IN) mask[i] = 1.0f;
}
__global__ void k_mask_scatter(const int* __restrict__ mg, float* mask) {
  int i = blockIdx.x * blockDim.x + threadIdx.x;
  if (i < BB) mask[mg[i]] = 0.0f;   // all writers store 0: race-free semantics
}

// --------------------- encoder embedding gather-matvec ---------------------
// One block per sample. 512 MB HBM stream: fully coalesced b128 loads.
__global__ void __launch_bounds__(256)
k_enc_gather(const float* __restrict__ x, const int* __restrict__ mg,
             const float* __restrict__ mask, const float* __restrict__ amats,
             const float* __restrict__ bvecs, _Float16* __restrict__ h0,
             float* __restrict__ library) {
  __shared__ __align__(16) float xs[N_IN];
  __shared__ float red[256];
  const int b = blockIdx.x, tid = threadIdx.x;

  float lib = 0.0f;
  for (int c = tid; c < N_IN; c += 256) {
    float xv = x[(size_t)b * N_IN + c];
    lib += xv;
    xs[c] = log1pf(xv * mask[c]);
  }
  red[tid] = lib;
  __syncthreads();
  for (int s = 128; s > 0; s >>= 1) {
    if (tid < s) red[tid] += red[tid + s];
    __syncthreads();
  }
  if (tid == 0) library[b] = red[0];
  // xs[] is visible to all threads (barriers above).

  const int g = mg[b];
  const float* Abase = amats + (size_t)g * (N_IN * HH);
  const float* bv    = bvecs + (size_t)g * HH;
  const int wave = tid >> 5, lane = tid & 31;
  const vf4* xs4 = (const vf4*)xs;

#pragma unroll 1
  for (int t = 0; t < 16; ++t) {
    const int r = wave + 8 * t;                       // 8 waves cover 128 rows
    const vf4* row4 = (const vf4*)(Abase + (size_t)r * N_IN);
    float acc = 0.0f;
    for (int i4 = lane; i4 < 250; i4 += 32) {         // 250 float4 = 1000 f32
      vf4 a = row4[i4];
      vf4 xv = xs4[i4];
      acc += a.x * xv.x + a.y * xv.y + a.z * xv.z + a.w * xv.w;
    }
    acc = wave_sum(acc);
    if (lane == 0) h0[(size_t)b * HH + r] = (_Float16)(acc + bv[r]);
  }
}

// ------------------ weight f32 -> f16 WMMA B-fragment swizzle --------------
// Fragment layout for V_WMMA_F32_16X16X32_F16 B operand (contiguous k, same order as load_fragA):
//   lane<16 : col=lane,    K = ks*32 + [0..15]
//   lane>=16: col=lane-16, K = ks*32 + [16..31]
// dst element index: (((nt*KS + ks)*32 + lane)*16 + i)
__global__ void k_swizzleB(const float* __restrict__ src, int K, int Nsrc,
                           _Float16* __restrict__ dst) {
  const int KS = K >> 5;
  const int ntiles = (Nsrc + 15) >> 4;
  const int total = ntiles * KS * 32;
  int gid = blockIdx.x * blockDim.x + threadIdx.x;
  if (gid >= total) return;
  const int lane = gid & 31;
  const int ks = (gid >> 5) % KS;
  const int nt = gid / (32 * KS);
  const int n = nt * 16 + (lane & 15);
  const int kbase = ks * 32 + (lane >> 4) * 16;
  _Float16* d = dst + (size_t)gid * 16;
#pragma unroll
  for (int i = 0; i < 16; ++i) {
    float v = (n < Nsrc) ? src[(size_t)(kbase + i) * Nsrc + n] : 0.0f;
    d[i] = (_Float16)v;
  }
}

// --------------------------- fragment loaders ------------------------------
__device__ __forceinline__ v16h load_fragA(const _Float16* __restrict__ A,
                                           int m0, int K, int ks, int r, int hi) {
  Frag a;
  const _Float16* ap = A + (size_t)(m0 + r) * K + ks * 32 + hi * 16;
  a.f[0] = *(const vf4*)(ap);        // K = ks*32 + hi*16 + 0..7
  a.f[1] = *(const vf4*)(ap + 8);    // K = ks*32 + hi*16 + 8..15
  return a.v;
}
__device__ __forceinline__ v16h load_fragB(const _Float16* __restrict__ Bf,
                                           int KS, int nt, int ks, int lane) {
  Frag b;
  const _Float16* bp = Bf + (((size_t)nt * KS + ks) * 32 + lane) * 16;
  b.f[0] = *(const vf4*)(bp);
  b.f[1] = *(const vf4*)(bp + 8);
  return b.v;
}

// ------------------------------ epilogue -----------------------------------
// Modes: 0 enc1(+bias, f32)  1 meanlv(+bias|bias2, f32)
//        2 dec1(+bias+w2[bidx]+b2, BN1, ReLU, f16)
//        3 dec3(+bias, BN2, ReLU, f16)  4 dec5(+bias on n<Nvalid, f32)
template <int MODE>
__device__ __forceinline__ void epilogue(
    v8f acc, int m0, int n0, int hi, int lane, int N,
    float* __restrict__ outF32, _Float16* __restrict__ outF16,
    const float* __restrict__ bias, const float* __restrict__ bias2,
    const float* __restrict__ bnscale, const float* __restrict__ bnbias,
    const float* __restrict__ bnmean, const float* __restrict__ bnvar,
    const float* __restrict__ batch_oh, const float* __restrict__ w2,
    const float* __restrict__ b2, int Nvalid) {
  const int n = n0 + (lane & 15);
#pragma unroll
  for (int i = 0; i < 8; ++i) {
    const int m = m0 + i + (hi << 3);        // D layout: VGPR i -> row i (+8 hi)
    float v = acc[i];
    if (MODE == 0) {
      outF32[(size_t)m * N + n] = v + bias[n];
    } else if (MODE == 1) {
      float bb = (n < LL) ? bias[n] : bias2[n - LL];
      outF32[(size_t)m * N + n] = v + bb;
    } else if (MODE == 2) {
      int bidx = (batch_oh[(size_t)m * 2 + 1] > batch_oh[(size_t)m * 2]) ? 1 : 0;
      float t = v + bias[n] + w2[bidx * HH + n] + b2[n];
      t = (t - bnmean[n]) * rsqrtf(bnvar[n] + BN_EPS) * bnscale[n] + bnbias[n];
      outF16[(size_t)m * N + n] = (_Float16)fmaxf(t, 0.0f);
    } else if (MODE == 3) {
      float t = v + bias[n];
      t = (t - bnmean[n]) * rsqrtf(bnvar[n] + BN_EPS) * bnscale[n] + bnbias[n];
      outF16[(size_t)m * N + n] = (_Float16)fmaxf(t, 0.0f);
    } else { // MODE 4
      outF32[(size_t)m * N + n] = v + ((n < Nvalid) ? bias[n] : 0.0f);
    }
  }
}

// --------------------------- WMMA GEMM (templated) -------------------------
// C[M,N] = A_f16[M,K] @ B_frag_f16[K,N]; one wave -> NTW adjacent 16x16 tiles
// sharing the A fragment; K loop fully unrolled (KS compile-time).
template <int KS, int MODE, int NTW>
__global__ void __launch_bounds__(128)
k_gemm_wmma(const _Float16* __restrict__ A, const _Float16* __restrict__ Bf,
            int M, int N, float* __restrict__ outF32,
            _Float16* __restrict__ outF16,
            const float* __restrict__ bias, const float* __restrict__ bias2,
            const float* __restrict__ bnscale, const float* __restrict__ bnbias,
            const float* __restrict__ bnmean, const float* __restrict__ bnvar,
            const float* __restrict__ batch_oh, const float* __restrict__ w2,
            const float* __restrict__ b2, int Nvalid) {
  const int wave = threadIdx.x >> 5;
  const int lane = threadIdx.x & 31;
  const int K = KS * 32;
  const int ntN = (N >> 4) / NTW;                 // tile-groups along N
  const int grpTot = (M >> 4) * ntN;
  const int grp = blockIdx.x * 4 + wave;
  if (grp >= grpTot) return;
  const int mt = grp / ntN, ng = grp % ntN;
  const int m0 = mt << 4;
  const int r = lane & 15, hi = lane >> 4;

  v8f acc[NTW];
#pragma unroll
  for (int j = 0; j < NTW; ++j) acc[j] = (v8f){};

#pragma unroll
  for (int ks = 0; ks < KS; ++ks) {
    const v16h a = load_fragA(A, m0, K, ks, r, hi);
#pragma unroll
    for (int j = 0; j < NTW; ++j) {
      const v16h bfrag = load_fragB(Bf, KS, ng * NTW + j, ks, lane);
      acc[j] = __builtin_amdgcn_wmma_f32_16x16x32_f16(
          false, a, false, bfrag, (short)0, acc[j], false, false);
    }
  }

#pragma unroll
  for (int j = 0; j < NTW; ++j) {
    epilogue<MODE>(acc[j], m0, (ng * NTW + j) << 4, hi, lane, N,
                   outF32, outF16, bias, bias2, bnscale, bnbias, bnmean, bnvar,
                   batch_oh, w2, b2, Nvalid);
  }
}

// --------------------------- LayerNorm + ReLU ------------------------------
__global__ void __launch_bounds__(128)
k_ln_relu(const float* __restrict__ H1, const float* __restrict__ scale,
          const float* __restrict__ bias, _Float16* __restrict__ out) {
  const int wave = threadIdx.x >> 5, lane = threadIdx.x & 31;
  const int b = blockIdx.x * 4 + wave;
  const float* row = H1 + (size_t)b * HH;
  float v[4];
  float s = 0.0f;
#pragma unroll
  for (int i = 0; i < 4; ++i) { v[i] = row[lane + 32 * i]; s += v[i]; }
  s = wave_sum(s);
  const float mu = s * (1.0f / HH);
  float q = 0.0f;
#pragma unroll
  for (int i = 0; i < 4; ++i) { float d = v[i] - mu; q += d * d; }
  q = wave_sum(q);
  const float inv = rsqrtf(q * (1.0f / HH) + LN_EPS);
#pragma unroll
  for (int i = 0; i < 4; ++i) {
    int c = lane + 32 * i;
    float t = (v[i] - mu) * inv * scale[c] + bias[c];
    out[(size_t)b * HH + c] = (_Float16)fmaxf(t, 0.0f);
  }
}

// ----------------------- reparameterize z + KL per row ---------------------
__global__ void __launch_bounds__(128)
k_latent(const float* __restrict__ ml, const float* __restrict__ znoise,
         _Float16* __restrict__ z, float* __restrict__ kl) {
  const int wave = threadIdx.x >> 5, lane = threadIdx.x & 31;
  const int b = blockIdx.x * 4 + wave;
  const float mean = ml[(size_t)b * 64 + lane];
  const float lv   = ml[(size_t)b * 64 + LL + lane];
  const float sigma = expf(lv);
  z[(size_t)b * LL + lane] = (_Float16)(mean + sigma * znoise[(size_t)b * LL + lane]);
  float k = -lv + 0.5f * (expf(2.0f * lv) + mean * mean - 1.0f);
  k = wave_sum(k);
  if (lane == 0) kl[b] = k;
}

// ------------------- softmax + Poisson log-lik per sample ------------------
__global__ void __launch_bounds__(256)
k_rowloss(const float* __restrict__ logits, const float* __restrict__ x,
          const float* __restrict__ library, const float* __restrict__ kl,
          float* __restrict__ elbo) {
  __shared__ float red[256];
  const int b = blockIdx.x, tid = threadIdx.x;
  const float* lr = logits + (size_t)b * N_INP;

  float mx = -1e30f;
  for (int c = tid; c < N_IN; c += 256) mx = fmaxf(mx, lr[c]);
  red[tid] = mx;
  __syncthreads();
  for (int s = 128; s > 0; s >>= 1) {
    if (tid < s) red[tid] = fmaxf(red[tid], red[tid + s]);
    __syncthreads();
  }
  mx = red[0];
  __syncthreads();

  float se = 0.0f;
  for (int c = tid; c < N_IN; c += 256) se += expf(lr[c] - mx);
  red[tid] = se;
  __syncthreads();
  for (int s = 128; s > 0; s >>= 1) {
    if (tid < s) red[tid] += red[tid + s];
    __syncthreads();
  }
  const float logZ = logf(red[0]);
  __syncthreads();

  const float lib = library[b];
  const float loglib = logf(lib);
  float acc = 0.0f;
  for (int c = tid; c < N_IN; c += 256) {
    float xv = x[(size_t)b * N_IN + c];
    float lograte = lr[c] - mx - logZ + loglib;
    acc += xv * lograte - lgammaf(xv + 1.0f);
  }
  red[tid] = acc;
  __syncthreads();
  for (int s = 128; s > 0; s >>= 1) {
    if (tid < s) red[tid] += red[tid + s];
    __syncthreads();
  }
  if (tid == 0) elbo[b] = red[0] - lib - kl[b];   // sum(rate) == library
}

// --------------------------- final mean reduce -----------------------------
__global__ void __launch_bounds__(256)
k_final(const float* __restrict__ elbo, float* __restrict__ out) {
  __shared__ float red[256];
  const int tid = threadIdx.x;
  float s = 0.0f;
  for (int i = tid; i < BB; i += 256) s += elbo[i];
  red[tid] = s;
  __syncthreads();
  for (int st = 128; st > 0; st >>= 1) {
    if (tid < st) red[tid] += red[tid + st];
    __syncthreads();
  }
  if (tid == 0) out[0] = -red[0] * (1.0f / BB);
}

// ---------------------------------------------------------------------------
static inline int cdiv(int a, int b) { return (a + b - 1) / b; }

extern "C" void kernel_launch(void* const* d_in, const int* in_sizes, int n_in,
                              void* d_out, int out_size, void* d_ws, size_t ws_size,
                              hipStream_t stream) {
  (void)in_sizes; (void)n_in; (void)out_size; (void)ws_size;
  // ---- inputs (setup_inputs dict order, params flattened in insertion order)
  const float* x        = (const float*)d_in[0];
  const float* batch_oh = (const float*)d_in[1];
  const int*   mg       = (const int*)  d_in[2];
  const float* z_noise  = (const float*)d_in[3];
  const float* amats    = (const float*)d_in[4];
  const float* bvecs    = (const float*)d_in[5];
  const float* enc_w1   = (const float*)d_in[6];
  const float* enc_b1   = (const float*)d_in[7];
  const float* ln_scale = (const float*)d_in[8];
  const float* ln_bias  = (const float*)d_in[9];
  const float* w_mean   = (const float*)d_in[10];
  const float* b_mean   = (const float*)d_in[11];
  const float* w_lv     = (const float*)d_in[12];
  const float* b_lv     = (const float*)d_in[13];
  const float* dec_w1   = (const float*)d_in[14];
  const float* dec_b1   = (const float*)d_in[15];
  const float* dec_w2   = (const float*)d_in[16];
  const float* dec_b2   = (const float*)d_in[17];
  const float* dec_w3   = (const float*)d_in[18];
  const float* dec_b3   = (const float*)d_in[19];
  const float* dec_w5   = (const float*)d_in[20];
  const float* dec_b5   = (const float*)d_in[21];
  const float* bn1_s    = (const float*)d_in[22];
  const float* bn1_b    = (const float*)d_in[23];
  const float* bn1_m    = (const float*)d_in[24];
  const float* bn1_v    = (const float*)d_in[25];
  const float* bn2_s    = (const float*)d_in[26];
  const float* bn2_b    = (const float*)d_in[27];
  const float* bn2_m    = (const float*)d_in[28];
  const float* bn2_v    = (const float*)d_in[29];
  float* out = (float*)d_out;

  // ---- workspace carve-up (256B aligned) ----
  char* ws = (char*)d_ws;
  size_t off = 0;
  auto alloc = [&](size_t bytes) { char* p = ws + off; off += (bytes + 255) & ~(size_t)255; return p; };
  float*    maskv   = (float*)   alloc(N_IN * 4);
  _Float16* h0f     = (_Float16*)alloc((size_t)BB * HH * 2);
  _Float16* fW1     = (_Float16*)alloc(8 * 4 * 32 * 16 * 2);        // enc_w1
  _Float16* fMLV    = (_Float16*)alloc(4 * 4 * 32 * 16 * 2);        // [mean|lv]
  _Float16* fDW1    = (_Float16*)alloc(8 * 1 * 32 * 16 * 2);        // dec_w1 (K=32)
  _Float16* fDW3    = (_Float16*)alloc(8 * 4 * 32 * 16 * 2);        // dec_w3
  _Float16* fDW5    = (_Float16*)alloc(63 * 4 * 32 * 16 * 2);       // dec_w5 (N pad 1008)
  float*    H1      = (float*)   alloc((size_t)BB * HH * 4);
  _Float16* Hl      = (_Float16*)alloc((size_t)BB * HH * 2);
  float*    ml      = (float*)   alloc((size_t)BB * 64 * 4);
  _Float16* zf      = (_Float16*)alloc((size_t)BB * LL * 2);
  float*    klv     = (float*)   alloc((size_t)BB * 4);
  _Float16* D0      = (_Float16*)alloc((size_t)BB * HH * 2);
  _Float16* D1      = (_Float16*)alloc((size_t)BB * HH * 2);
  float*    logits  = (float*)   alloc((size_t)BB * N_INP * 4);
  float*    library = (float*)   alloc((size_t)BB * 4);
  float*    elbo    = (float*)   alloc((size_t)BB * 4);

  // ---- global gene mask (union of all masked_genes columns) ----
  k_mask_init<<<cdiv(N_IN, 256), 256, 0, stream>>>(maskv);
  k_mask_scatter<<<cdiv(BB, 256), 256, 0, stream>>>(mg, maskv);

  // ---- weight fragment conversion (f32 -> f16, WMMA B layout) ----
  k_swizzleB<<<cdiv(8 * 4 * 32, 256), 256, 0, stream>>>(enc_w1, 128, 128, fW1);
  k_swizzleB<<<cdiv(2 * 4 * 32, 256), 256, 0, stream>>>(w_mean, 128, 32, fMLV);
  k_swizzleB<<<cdiv(2 * 4 * 32, 256), 256, 0, stream>>>(w_lv,   128, 32, fMLV + 2 * 4 * 32 * 16);
  k_swizzleB<<<cdiv(8 * 1 * 32, 256), 256, 0, stream>>>(dec_w1, 32, 128, fDW1);
  k_swizzleB<<<cdiv(8 * 4 * 32, 256), 256, 0, stream>>>(dec_w3, 128, 128, fDW3);
  k_swizzleB<<<cdiv(63 * 4 * 32, 256), 256, 0, stream>>>(dec_w5, 128, 1000, fDW5);

  // ---- encoder embedding gather (the 512 MB HBM stream) ----
  k_enc_gather<<<BB, 256, 0, stream>>>(x, mg, maskv, amats, bvecs, h0f, library);

  // ---- H1 = h0 @ enc_w1 + b1 (WMMA, K=128) ----
  k_gemm_wmma<4, 0, 1><<<(BB / 16) * (HH / 16) / 4, 128, 0, stream>>>(
      h0f, fW1, BB, HH, H1, nullptr, enc_b1,
      nullptr, nullptr, nullptr, nullptr, nullptr, nullptr, nullptr, nullptr, HH);

  // ---- LN + ReLU -> Hl (f16) ----
  k_ln_relu<<<BB / 4, 128, 0, stream>>>(H1, ln_scale, ln_bias, Hl);

  // ---- [mean|logvar] = Hl @ [w_mean|w_lv] (WMMA, K=128) ----
  k_gemm_wmma<4, 1, 1><<<(BB / 16) * (64 / 16) / 4, 128, 0, stream>>>(
      Hl, fMLV, BB, 64, ml, nullptr, b_mean,
      b_lv, nullptr, nullptr, nullptr, nullptr, nullptr, nullptr, nullptr, 64);

  // ---- z = mean + exp(lv)*noise ; KL per sample ----
  k_latent<<<BB / 4, 128, 0, stream>>>(ml, z_noise, zf, klv);

  // ---- D0 = relu(bn1(z @ dec_w1 + b1 + dec_w2[bidx] + b2)) (WMMA, K=32) ----
  k_gemm_wmma<1, 2, 1><<<(BB / 16) * (HH / 16) / 4, 128, 0, stream>>>(
      zf, fDW1, BB, HH, nullptr, D0, dec_b1,
      nullptr, bn1_s, bn1_b, bn1_m, bn1_v, batch_oh, dec_w2, dec_b2, HH);

  // ---- D1 = relu(bn2(D0 @ dec_w3 + b3)) (WMMA, K=128) ----
  k_gemm_wmma<4, 3, 1><<<(BB / 16) * (HH / 16) / 4, 128, 0, stream>>>(
      D0, fDW3, BB, HH, nullptr, D1, dec_b3,
      nullptr, bn2_s, bn2_b, bn2_m, bn2_v, nullptr, nullptr, nullptr, HH);

  // ---- logits = D1 @ dec_w5 + b5 (WMMA, K=128, N=1008, 3 N-tiles/wave) ----
  // 63 N-tiles = 21 groups of 3; A fragment reused across the 3 tiles.
  k_gemm_wmma<4, 4, 3><<<cdiv((BB / 16) * 21, 4), 128, 0, stream>>>(
      D1, fDW5, BB, N_INP, logits, nullptr, dec_b5,
      nullptr, nullptr, nullptr, nullptr, nullptr, nullptr, nullptr, nullptr, N_IN);

  // ---- softmax + Poisson log-lik per sample, then final mean ----
  k_rowloss<<<BB, 256, 0, stream>>>(logits, x, library, klv, elbo);
  k_final<<<1, 256, 0, stream>>>(elbo, out);
}
